// C2QAttention_76441827934562
// MI455X (gfx1250) — matrix operand order repaired and checked by reading, not compiled, not run on hardware
//
#include <hip/hip_runtime.h>
#include <stdint.h>

typedef __attribute__((ext_vector_type(16))) __bf16 bf16x16;
typedef __attribute__((ext_vector_type(8)))  float  f32x8;
typedef __attribute__((ext_vector_type(4)))  unsigned int u32x4;
typedef __attribute__((ext_vector_type(8)))  int i32x8;
typedef __attribute__((ext_vector_type(4)))  int i32x4;

#define NEG_INF_F (-1e30f)

__device__ __forceinline__ f32x8 wmma_bf16(bf16x16 a, bf16x16 b, f32x8 c) {
  return __builtin_amdgcn_wmma_f32_16x16x32_bf16(false, a, false, b, (short)0, c, false, false);
}

// Low 32 bits of a generic pointer to __shared__ data = LDS byte offset.
__device__ __forceinline__ unsigned lds_off(const void* p) {
  return (unsigned)(uintptr_t)p;
}

// TDM: 2-D f32 tile DMA, global -> LDS.  D# per CDNA5 ISA ch.8:
//   group0: count=1, lds_addr[63:32], global_addr[120:64], type=2 [127:126]
//   group1: data_size=2 (4B) [17:16]; tensor_dim0 [79:48]; tensor_dim1 [111:80];
//           tile_dim0 [127:112]; tile_dim1 [143:128]; dim0_stride [207:160]
//   groups 2/3 + extended group: zero (2-D tensor)
// 6-arg builtin form (clang-23 / therock-10.0 headers).
__device__ __forceinline__ void tdm_load_2d_f32(unsigned lds_byte_off, const float* gptr,
                                                unsigned dim0, unsigned dim1,
                                                unsigned tile0, unsigned tile1,
                                                unsigned long long stride0) {
  unsigned long long ga = (unsigned long long)(uintptr_t)gptr;
  u32x4 g0;
  g0[0] = 1u;                                               // count=1
  g0[1] = lds_byte_off;                                     // lds_addr
  g0[2] = (unsigned)(ga & 0xffffffffu);                     // global_addr lo
  g0[3] = (unsigned)((ga >> 32) & 0x01ffffffu) | (2u << 30);// global_addr hi | type=2
  i32x8 g1;
  g1[0] = (int)(2u << 16);                                  // data_size = 4 bytes
  g1[1] = (int)((dim0 & 0xffffu) << 16);                    // tensor_dim0[15:0]
  g1[2] = (int)(((dim0 >> 16) & 0xffffu) | ((dim1 & 0xffffu) << 16));
  g1[3] = (int)(((dim1 >> 16) & 0xffffu) | ((tile0 & 0xffffu) << 16));
  g1[4] = (int)(tile1 & 0xffffu);                           // tile_dim1, tile_dim2=0
  g1[5] = (int)(unsigned)(stride0 & 0xffffffffu);           // dim0_stride lo
  g1[6] = (int)(unsigned)((stride0 >> 32) & 0xffffu);       // dim0_stride hi, dim1_stride=0
  g1[7] = 0;
  i32x4 gz4 = {0, 0, 0, 0};
  i32x8 gz8 = {0, 0, 0, 0, 0, 0, 0, 0};
  __builtin_amdgcn_tensor_load_to_lds(g0, g1, gz4, gz4, gz8, 0);
}

// ---------------------------------------------------------------- kernel 1
// s1q[b][j] = dot(q[b,j,:], q_weight) + bias
__global__ void __launch_bounds__(128)
k_s1q(const float* __restrict__ q, const float* __restrict__ qw,
      const float* __restrict__ bias, float* __restrict__ s1q) {
  const int b = blockIdx.x, j = threadIdx.x;
  const float* qr = q + ((size_t)b * 128 + j) * 128;
  float acc = bias[0];
#pragma unroll 8
  for (int d = 0; d < 128; ++d) acc += qr[d] * qw[d];
  s1q[b * 128 + j] = acc;
}

// ---------------------------------------------------------------- kernel 2
// Per (i-tile of 64 rows, batch): S = s0 + s1 + (c*cqw)@q^T, row softmax -> S1,
// A = S1 @ q.  Writes S1 to ws, out[:,0:128]=c, [128:256]=A, [256:384]=c*A.
__global__ void __launch_bounds__(128)
k_rows(const float* __restrict__ c, const float* __restrict__ q,
       const int* __restrict__ qmask, const float* __restrict__ cweight,
       const float* __restrict__ cqw_g, const float* __restrict__ s1q_g,
       float* __restrict__ S1ws, float* __restrict__ out) {
  __shared__ float  Qf32[128 * 128];         // TDM destination (q tile, f32)
  __shared__ __bf16 Qbf[128 * 128];          // q staged bf16 (row-major j,d)
  __shared__ float  cw_lds[128], cqw_lds[128], s1q_lds[128], qm_lds[128];
  __shared__ float  s0_lds[64];
  __shared__ __bf16 slab[4][16 * 128];       // per-wave S1 slab (bf16)

  const int b    = blockIdx.y;
  const int i0   = blockIdx.x * 64;
  const int tid  = threadIdx.x;
  const int w    = tid >> 5;
  const int lane = tid & 31;
  const int half = lane >> 4;
  const int ln   = lane & 15;

  // ---- TDM: DMA whole q[b] tile (128x128 f32) into LDS; wave 0 issues.
  const float* qb = q + (size_t)b * 128 * 128;
  if (w == 0) {
    tdm_load_2d_f32(lds_off(Qf32), qb, 128, 128, 128, 128, 128);
    __builtin_amdgcn_s_wait_tensorcnt(0);
  }
  cw_lds[tid]  = cweight[tid];
  cqw_lds[tid] = cqw_g[tid];
  s1q_lds[tid] = s1q_g[b * 128 + tid];
  qm_lds[tid]  = (float)qmask[b * 128 + tid];
  __syncthreads();

  // ---- convert q to bf16; s0[i] = c[i,:] . c_weight
  for (int idx = tid; idx < 128 * 128; idx += 128) Qbf[idx] = (__bf16)Qf32[idx];
  if (tid < 64) {
    const float* cr = c + ((size_t)b * 1024 + i0 + tid) * 128;
    float a = 0.f;
#pragma unroll 8
    for (int d = 0; d < 128; ++d) a += cr[d] * cw_lds[d];
    s0_lds[tid] = a;
  }
  __syncthreads();

  const int rowbase = i0 + w * 16;

  // ---- GEMM1: A-frags = (c * cq_weight) rows, bf16
  const float* crow = c + ((size_t)b * 1024 + rowbase + ln) * 128;
  bf16x16 afr[4];
#pragma unroll
  for (int kc = 0; kc < 4; ++kc) {
    const int kb = kc * 32 + (half ? 8 : 0);
#pragma unroll
    for (int t = 0; t < 8; ++t) {
      afr[kc][t]     = (__bf16)(crow[kb + t]      * cqw_lds[kb + t]);
      afr[kc][8 + t] = (__bf16)(crow[kb + 16 + t] * cqw_lds[kb + 16 + t]);
    }
  }

  f32x8 acc[8] = {};
#pragma unroll
  for (int nt = 0; nt < 8; ++nt) {
    const int j = nt * 16 + ln;
#pragma unroll
    for (int kc = 0; kc < 4; ++kc) {
      const int kb = kc * 32 + (half ? 8 : 0);
      bf16x16 bfr;
#pragma unroll
      for (int i = 0; i < 16; ++i) bfr[i] = Qbf[j * 128 + kb + (i < 8 ? i : i + 8)];  // q[j][kb+(i<8?i:i+8)]
      acc[nt] = wmma_bf16(afr[kc], bfr, acc[nt]);
    }
  }

  // ---- masked row softmax, entirely in fragments (rows = reg idx + lane half)
  float sval[8][8];
  float rmax[8];
#pragma unroll
  for (int r = 0; r < 8; ++r) rmax[r] = NEG_INF_F;
#pragma unroll
  for (int nt = 0; nt < 8; ++nt) {
    const int   j   = nt * 16 + ln;
    const float s1j = s1q_lds[j];
    const float qm  = qm_lds[j];
#pragma unroll
    for (int r = 0; r < 8; ++r) {
      float v = acc[nt][r] + s0_lds[w * 16 + r + (half ? 8 : 0)] + s1j;
      v = (qm != 0.f) ? v : NEG_INF_F;
      sval[nt][r] = v;
      rmax[r] = fmaxf(rmax[r], v);
    }
  }
#pragma unroll
  for (int r = 0; r < 8; ++r) {
    float m = rmax[r];
    for (int off = 1; off < 16; off <<= 1) m = fmaxf(m, __shfl_xor(m, off, 32));
    rmax[r] = m;
  }
  float rsum[8] = {};
#pragma unroll
  for (int nt = 0; nt < 8; ++nt)
#pragma unroll
    for (int r = 0; r < 8; ++r) {
      const float p = __expf(sval[nt][r] - rmax[r]);
      sval[nt][r] = p;
      rsum[r] += p;
    }
#pragma unroll
  for (int r = 0; r < 8; ++r) {
    float s = rsum[r];
    for (int off = 1; off < 16; off <<= 1) s += __shfl_xor(s, off, 32);
    rsum[r] = 1.f / s;  // >= 1 always (max element contributes exp(0)=1)
  }

  // ---- write S1 (f32 ws + bf16 per-wave slab)
  __bf16* myslab = slab[w];
#pragma unroll
  for (int nt = 0; nt < 8; ++nt) {
#pragma unroll
    for (int r = 0; r < 8; ++r) {
      const int   lr  = r + (half ? 8 : 0);
      const float s1v = sval[nt][r] * rsum[r];
      S1ws[((size_t)b * 1024 + rowbase + lr) * 128 + nt * 16 + ln] = s1v;
      myslab[lr * 128 + nt * 16 + ln] = (__bf16)s1v;
    }
  }
  __syncthreads();

  // ---- GEMM2: A = S1 @ q
  bf16x16 a2[4];
#pragma unroll
  for (int kc = 0; kc < 4; ++kc) {
    const int kb = kc * 32 + (half ? 8 : 0);
#pragma unroll
    for (int t = 0; t < 8; ++t) {
      a2[kc][t]     = myslab[ln * 128 + kb + t];
      a2[kc][8 + t] = myslab[ln * 128 + kb + 16 + t];
    }
  }
  f32x8 acc2[8] = {};
#pragma unroll
  for (int nt = 0; nt < 8; ++nt) {
    const int d = nt * 16 + ln;
#pragma unroll
    for (int kc = 0; kc < 4; ++kc) {
      const int kb = kc * 32 + (half ? 8 : 0);
      bf16x16 bfr;
#pragma unroll
      for (int i = 0; i < 16; ++i) bfr[i] = Qbf[(kb + (i < 8 ? i : i + 8)) * 128 + d];  // q[k][d]
      acc2[nt] = wmma_bf16(a2[kc], bfr, acc2[nt]);
    }
  }

  // ---- out[:,0:128]=c, [128:256]=A, [256:384]=c*A
#pragma unroll
  for (int nt = 0; nt < 8; ++nt) {
    const int d = nt * 16 + ln;
#pragma unroll
    for (int r = 0; r < 8; ++r) {
      const int    row   = rowbase + r + (half ? 8 : 0);
      const size_t cidx  = ((size_t)b * 1024 + row) * 128 + d;
      const size_t obase = ((size_t)b * 1024 + row) * 512;
      const float  cv = c[cidx];
      const float  av = acc2[nt][r];
      out[obase + d]       = cv;
      out[obase + 128 + d] = av;
      out[obase + 256 + d] = cv * av;
    }
  }
}

// ---------------------------------------------------------------- kernel 3
// Per batch: T[j,d] = (1/Z[j]) * sum_k cm[k]*exp(S1[k,j])*c[k,d]
// (column softmax of S1 needs no max pass since S1 in [0,1]).
// TDM double-buffers the 64-row K-panels of S1 and c.
// Writes T transposed (d-major) for kernel 4's contiguous B-frags.
__global__ void __launch_bounds__(256)
k_colT(const float* __restrict__ c, const int* __restrict__ cmask,
       const float* __restrict__ S1ws, float* __restrict__ TTws) {
  __shared__ float  S1p[2][64 * 128];  // TDM dest: S1 panel (f32)
  __shared__ float  Cp[2][64 * 128];   // TDM dest: c panel (f32)
  __shared__ __bf16 Elds[64 * 128];    // E[k][j] bf16 panel
  __shared__ __bf16 cT[128 * 66];      // c^T[d][k] bf16 panel (stride 66: no conflicts)
  __shared__ float  Zlds[128];

  const int b    = blockIdx.x;
  const int tid  = threadIdx.x;
  const int w    = tid >> 5;
  const int lane = tid & 31;
  const int half = lane >> 4;
  const int ln   = lane & 15;

  const float* S1b = S1ws + (size_t)b * 1024 * 128;
  const float* cb  = c    + (size_t)b * 1024 * 128;

  if (tid < 128) Zlds[tid] = 0.f;

  // prologue: DMA panel 0 into buffer 0
  if (w == 0) {
    tdm_load_2d_f32(lds_off(S1p[0]), S1b, 128, 64, 128, 64, 128);
    tdm_load_2d_f32(lds_off(Cp[0]),  cb,  128, 64, 128, 64, 128);
  }

  f32x8 acc[8] = {};
  for (int p = 0; p < 16; ++p) {
    const int s = p & 1;
    if (w == 0) {
      if (p < 15) {  // prefetch next panel, then wait for current pair only
        tdm_load_2d_f32(lds_off(S1p[s ^ 1]), S1b + (size_t)(p + 1) * 64 * 128,
                        128, 64, 128, 64, 128);
        tdm_load_2d_f32(lds_off(Cp[s ^ 1]),  cb  + (size_t)(p + 1) * 64 * 128,
                        128, 64, 128, 64, 128);
        __builtin_amdgcn_s_wait_tensorcnt(2);
      } else {
        __builtin_amdgcn_s_wait_tensorcnt(0);
      }
    }
    __syncthreads();  // panel p resident; previous panel's E/cT fully consumed

    const int k0 = p * 64;
    for (int idx = tid; idx < 64 * 128; idx += 256) {
      const int   k  = idx >> 7, jd = idx & 127;
      const float s1 = S1p[s][idx];
      const float e  = cmask[b * 1024 + k0 + k] ? __expf(s1) : 0.f;
      Elds[idx] = (__bf16)e;
      atomicAdd(&Zlds[jd], e);
      cT[jd * 66 + k] = (__bf16)Cp[s][idx];   // jd plays 'd' for c
    }
    __syncthreads();

#pragma unroll
    for (int kc = 0; kc < 2; ++kc) {
      const int kb = kc * 32 + (half ? 8 : 0);
      bf16x16 afr;  // A[j][k] = E[k][j], rows j = w*16+ln
#pragma unroll
      for (int t = 0; t < 8; ++t) {
        afr[t]     = Elds[(kb + t) * 128 + w * 16 + ln];
        afr[8 + t] = Elds[(kb + 16 + t) * 128 + w * 16 + ln];
      }
      const int kb2 = kc * 32 + (half ? 8 : 0);
#pragma unroll
      for (int nt = 0; nt < 8; ++nt) {
        const int d = nt * 16 + ln;
        bf16x16 bfr;  // B[k][d] = c[k][d] = cT[d][k]
#pragma unroll
        for (int i = 0; i < 16; ++i) bfr[i] = cT[d * 66 + kb2 + (i < 8 ? i : i + 8)];
        acc[nt] = wmma_bf16(afr, bfr, acc[nt]);
      }
    }
    __syncthreads();
  }

#pragma unroll
  for (int nt = 0; nt < 8; ++nt) {
    const int d = nt * 16 + ln;
#pragma unroll
    for (int r = 0; r < 8; ++r) {
      const int   j = w * 16 + r + (half ? 8 : 0);
      const float z = Zlds[j];
      TTws[((size_t)b * 128 + d) * 128 + j] = acc[nt][r] / fmaxf(z, 1e-30f);
    }
  }
}

// ---------------------------------------------------------------- kernel 4
// Bm = S1 @ T ; out[:,384:512] = c * Bm
__global__ void __launch_bounds__(128)
k_bm(const float* __restrict__ c, const float* __restrict__ S1ws,
     const float* __restrict__ TTws, float* __restrict__ out) {
  __shared__ float  TTf32[128 * 128]; // TDM dest: T^T (d-major, f32)
  __shared__ __bf16 TTbf[128 * 128];  // T^T bf16
  __shared__ __bf16 slab[4][16 * 128];

  const int b    = blockIdx.y;
  const int i0   = blockIdx.x * 64;
  const int tid  = threadIdx.x;
  const int w    = tid >> 5;
  const int lane = tid & 31;
  const int half = lane >> 4;
  const int ln   = lane & 15;

  // ---- TDM: DMA T^T[b] tile (128x128 f32) into LDS
  if (w == 0) {
    tdm_load_2d_f32(lds_off(TTf32), TTws + (size_t)b * 128 * 128,
                    128, 128, 128, 128, 128);
    __builtin_amdgcn_s_wait_tensorcnt(0);
  }
  const int rowbase = i0 + w * 16;
  __bf16* myslab = slab[w];
  for (int idx = lane; idx < 16 * 128; idx += 32) {
    const int r = idx >> 7, j = idx & 127;
    myslab[idx] = (__bf16)S1ws[((size_t)b * 1024 + rowbase + r) * 128 + j];
  }
  __syncthreads();
  for (int idx = tid; idx < 128 * 128; idx += 128) TTbf[idx] = (__bf16)TTf32[idx];
  __syncthreads();

  bf16x16 a2[4];
#pragma unroll
  for (int kc = 0; kc < 4; ++kc) {
    const int kb = kc * 32 + (half ? 8 : 0);
#pragma unroll
    for (int t = 0; t < 8; ++t) {
      a2[kc][t]     = myslab[ln * 128 + kb + t];
      a2[kc][8 + t] = myslab[ln * 128 + kb + 16 + t];
    }
  }
  f32x8 acc[8] = {};
#pragma unroll
  for (int nt = 0; nt < 8; ++nt) {
    const int d = nt * 16 + ln;
#pragma unroll
    for (int kc = 0; kc < 4; ++kc) {
      const int kb = kc * 32 + (half ? 8 : 0);
      bf16x16 bfr;  // B[j][d] = T[j][d] = TTbf[d*128 + j], contiguous in j
#pragma unroll
      for (int i = 0; i < 16; ++i) bfr[i] = TTbf[d * 128 + kb + (i < 8 ? i : i + 8)];
      acc[nt] = wmma_bf16(a2[kc], bfr, acc[nt]);
    }
  }
#pragma unroll
  for (int nt = 0; nt < 8; ++nt) {
    const int d = nt * 16 + ln;
#pragma unroll
    for (int r = 0; r < 8; ++r) {
      const int   row = rowbase + r + (half ? 8 : 0);
      const float cv  = c[((size_t)b * 1024 + row) * 128 + d];
      out[((size_t)b * 1024 + row) * 512 + 384 + d] = cv * acc[nt][r];
    }
  }
}

// ---------------------------------------------------------------- launch
extern "C" void kernel_launch(void* const* d_in, const int* in_sizes, int n_in,
                              void* d_out, int out_size, void* d_ws, size_t ws_size,
                              hipStream_t stream) {
  (void)in_sizes; (void)n_in; (void)out_size; (void)ws_size;
  const float* c    = (const float*)d_in[0];
  const float* q    = (const float*)d_in[1];
  const int*   cm   = (const int*)d_in[2];
  const int*   qm   = (const int*)d_in[3];
  const float* cw   = (const float*)d_in[4];
  const float* qw   = (const float*)d_in[5];
  const float* cqw  = (const float*)d_in[6];
  const float* bias = (const float*)d_in[7];
  float* out = (float*)d_out;

  float* S1ws = (float*)d_ws;                       // 64*1024*128 f32 = 33.5 MB
  float* TTws = S1ws + (size_t)64 * 1024 * 128;     // 64*128*128  f32 = 4 MB
  float* s1q  = TTws + (size_t)64 * 128 * 128;      // 64*128      f32

  k_s1q <<<dim3(64),     128, 0, stream>>>(q, qw, bias, s1q);
  k_rows<<<dim3(16, 64), 128, 0, stream>>>(c, q, qm, cw, cqw, s1q, S1ws, out);
  k_colT<<<dim3(64),     256, 0, stream>>>(c, cm, S1ws, TTws);
  k_bm  <<<dim3(16, 64), 128, 0, stream>>>(c, S1ws, TTws, out);
}
